// Encoder_59871844107125
// MI455X (gfx1250) — matrix operand order repaired and checked by reading, not compiled, not run on hardware
//
#include <hip/hip_runtime.h>
#include <hip/hip_bf16.h>

typedef __attribute__((ext_vector_type(16))) __bf16 v16bf;
typedef __attribute__((ext_vector_type(8)))  __bf16 v8bf;
typedef __attribute__((ext_vector_type(4)))  __bf16 v4bf;
typedef __attribute__((ext_vector_type(8)))  float  v8f;

#define B_    32
#define T_IN  1024
#define F_    80
#define CU_   256
#define GU_   512
#define KW_   6
#define TOUT  512
#define G3    1536      // 3*GU
#define KCONV 480       // KW*F
#define WROWS 260       // 2*128 + KW - 2 : conv input window rows per 128-row tile

// ---------------- workspace layout (bytes, all 256-aligned) ----------------
constexpr size_t WS_Y      = 0;                                          // y bf16 [B][TOUT][CU]
constexpr size_t WS_XP     = WS_Y      + (size_t)B_*TOUT*CU_*2;          // xp bf16 [2][TOUT][B][G3]
constexpr size_t WS_WTCONV = WS_XP     + (size_t)2*TOUT*B_*G3*2;         // conv W^T bf16 [CU][KCONV]
constexpr size_t WS_KT     = WS_WTCONV + (size_t)CU_*KCONV*2;            // K^T bf16 [2][G3][CU]
constexpr size_t WS_HF     = WS_KT     + (size_t)2*G3*CU_*2;             // h f32 [2][B][GU]
constexpr size_t WS_HB     = WS_HF     + (size_t)2*B_*GU_*4;             // h bf16 [2][2 phase][B][GU]
constexpr size_t WS_MASK   = WS_HB     + (size_t)2*2*B_*GU_*2;           // mask f32 [B][TOUT]
constexpr size_t WS_CNT    = WS_MASK   + (size_t)B_*TOUT*4;              // counters [2][TOUT]

// ---------------- fragment helpers (wave32 WMMA bf16 layouts) ----------------
// A 16x32 bf16: lane<16 -> M=lane, K={kb..kb+7} U {kb+16..kb+23}, kb=kchunk
//               lane>=16 -> M=lane-16, kb=kchunk+8
__device__ inline v16bf load_a_frag(const __bf16* rowptr, int kchunk, int lane) {
  int kb = kchunk + ((lane >> 4) << 3);
  v8bf lo = *(const v8bf*)(rowptr + kb);
  v8bf hi = *(const v8bf*)(rowptr + kb + 16);
  v16bf a;
#pragma unroll
  for (int i = 0; i < 8; i++) { a[i] = lo[i]; a[i + 8] = hi[i]; }
  return a;
}
// B 32x16 bf16 from an [N][K] (transposed) row: lane<16 -> N=lane, K={kb..kb+7} U {kb+16..kb+23}, kb=kchunk
//                                               lane>=16 -> kb=kchunk+8
__device__ inline v16bf load_b_frag(const __bf16* rowptr, int kchunk, int lane) {
  int kb = kchunk + ((lane >> 4) << 3);
  v8bf lo = *(const v8bf*)(rowptr + kb);
  v8bf hi = *(const v8bf*)(rowptr + kb + 16);
  v16bf b;
#pragma unroll
  for (int i = 0; i < 8; i++) { b[i] = lo[i]; b[i + 8] = hi[i]; }
  return b;
}

__device__ inline float sigmoidf_(float x) { return 1.f / (1.f + __expf(-x)); }

// ---------------- kernels ----------------
__global__ void k_init(float* hf, __bf16* hb, int* cnt) {
  int i = blockIdx.x * blockDim.x + threadIdx.x;       // 65536 threads
  if (i < 2 * TOUT)      cnt[i] = 0;
  if (i < 2 * B_ * GU_)  hf[i]  = 0.f;
  if (i < 2 * 2 * B_ * GU_) hb[i] = (__bf16)0.f;
}

__global__ void k_prep(const float* __restrict__ conv_k,
                       const float* __restrict__ fK, const float* __restrict__ bK,
                       __bf16* __restrict__ wT, __bf16* __restrict__ kT) {
  int total = CU_ * KCONV + 2 * G3 * CU_;
  for (int i = blockIdx.x * blockDim.x + threadIdx.x; i < total; i += gridDim.x * blockDim.x) {
    if (i < CU_ * KCONV) {
      int n = i / KCONV, k = i - n * KCONV;
      wT[i] = (__bf16)conv_k[(size_t)k * CU_ + n];     // conv_k is [k=(kw,f)][cu]
    } else {
      int j = i - CU_ * KCONV;
      int d = j / (G3 * CU_); j -= d * (G3 * CU_);
      int n = j / CU_, k = j - n * CU_;
      const float* K = d ? bK : fK;                    // [CU][G3]
      kT[(size_t)d * G3 * CU_ + (size_t)n * CU_ + k] = (__bf16)K[(size_t)k * G3 + n];
    }
  }
}

__global__ void k_mask(const float* __restrict__ x, float* __restrict__ maskf,
                       float* __restrict__ outm) {
  int i = blockIdx.x * blockDim.x + threadIdx.x;       // b*TOUT + t
  if (i >= B_ * TOUT) return;
  int b = i >> 9, t = i & 511;
  const float* row = x + ((size_t)b * T_IN + 2 * t) * F_;
  float any = 0.f;
  for (int f = 0; f < F_; f++) { if (row[f] != 0.f) { any = 1.f; break; } }
  maskf[i] = any;
  outm[i]  = any;
}

// Conv as implicit GEMM: M=16384 (b,t), K=480 (kw,f), N=256.
// Block = 128x32 output tile (one batch), x window LDS-staged as bf16.
// Since 8 | 80, every 8-wide K-run of an A fragment sits inside one kw tap:
// A fragment = two aligned ds_load_b128, branch-free inner loop.
__global__ void __launch_bounds__(256) k_conv(const float* __restrict__ x,
                                              const __bf16* __restrict__ wT,
                                              const float* __restrict__ cb,
                                              __bf16* __restrict__ y) {
  __shared__ __bf16 ldsX[WROWS * F_];                  // 41.6 KB
  int bm = (int)blockIdx.x >> 3;                       // 128 M-blocks of 128 rows
  int bn = (int)blockIdx.x & 7;                        // 8 N-blocks of 32 cols
  int m0 = bm * 128;
  int b  = m0 >> 9;
  int t0 = m0 & 511;
  int n0 = bn * 32;
  int win = 2 * t0 - 2;                                // SAME pad_left = 2

  const float* xb = x + (size_t)b * T_IN * F_;
  for (int i = threadIdx.x; i < WROWS * (F_ / 4); i += 256) {
    int r  = i / (F_ / 4);
    int c4 = (i - r * (F_ / 4)) * 4;
    int tin = win + r;
    float4 v = make_float4(0.f, 0.f, 0.f, 0.f);
    if (tin >= 0 && tin < T_IN) v = *(const float4*)(xb + (size_t)tin * F_ + c4);
    v4bf o; o[0] = (__bf16)v.x; o[1] = (__bf16)v.y; o[2] = (__bf16)v.z; o[3] = (__bf16)v.w;
    *(v4bf*)(ldsX + r * F_ + c4) = o;
  }
  __syncthreads();

  int wave = (int)threadIdx.x >> 5, lane = (int)threadIdx.x & 31;
  int ttr = wave * 16 + (lane & 15);                   // local t of this lane's A row
  int na  = n0 + (lane & 15);
  const __bf16* browA = wT + (size_t)na * KCONV;
  const __bf16* browB = wT + (size_t)(na + 16) * KCONV;
  v8f acc0 = {}, acc1 = {};
  for (int kc = 0; kc < 15; kc++) {
    int kchunk = kc * 32;
    int kb = kchunk + ((lane >> 4) << 3);
    int kw1 = kb / F_,        f1 = kb - kw1 * F_;
    int kw2 = (kb + 16) / F_, f2 = (kb + 16) - kw2 * F_;
    v8bf lo = *(const v8bf*)(ldsX + (2 * ttr + kw1) * F_ + f1);
    v8bf hi = *(const v8bf*)(ldsX + (2 * ttr + kw2) * F_ + f2);
    v16bf a;
#pragma unroll
    for (int i = 0; i < 8; i++) { a[i] = lo[i]; a[i + 8] = hi[i]; }
    v16bf b0 = load_b_frag(browA, kchunk, lane);
    v16bf b1 = load_b_frag(browB, kchunk, lane);
    acc0 = __builtin_amdgcn_wmma_f32_16x16x32_bf16(false, a, false, b0, (short)0, acc0,
                                                   false, false);
    acc1 = __builtin_amdgcn_wmma_f32_16x16x32_bf16(false, a, false, b1, (short)0, acc1,
                                                   false, false);
  }
  float biasA = cb[na], biasB = cb[na + 16];
#pragma unroll
  for (int i = 0; i < 8; i++) {
    int tt = t0 + wave * 16 + i + ((lane >> 4) << 3);
    size_t base = ((size_t)b * TOUT + tt) * CU_;
    float v0 = acc0[i] + biasA; v0 = v0 > 0.f ? v0 : 0.f;
    float v1 = acc1[i] + biasB; v1 = v1 > 0.f ? v1 : 0.f;
    y[base + na]      = (__bf16)v0;
    y[base + na + 16] = (__bf16)v1;
  }
}

// xp = y @ K + b0, output time-major [t][b][g] bf16. M=16384 (t,b), K=256, N=1536.
__global__ void __launch_bounds__(256) k_proj(const __bf16* __restrict__ y,
                                              const __bf16* __restrict__ kT,
                                              const float* __restrict__ fb,
                                              const float* __restrict__ bb_,
                                              __bf16* __restrict__ xp) {
  int dir  = blockIdx.y;
  int wid  = ((int)blockIdx.x * 256 + (int)threadIdx.x) >> 5;   // 98304 waves
  int lane = (int)threadIdx.x & 31;
  int mtile = wid / 96, ntile = wid - mtile * 96;
  int m0 = mtile << 4, n0 = ntile << 4;
  const __bf16* kTd  = kT + (size_t)dir * G3 * CU_;
  const float*  bias = dir ? bb_ : fb;                          // row 0
  __bf16* xpd = xp + (size_t)dir * TOUT * B_ * G3;
  int mrow = m0 + (lane & 15);
  int t = mrow >> 5, b = mrow & 31;                             // m = t*32 + b
  const __bf16* arow = y + ((size_t)b * TOUT + t) * CU_;
  const __bf16* brow = kTd + (size_t)(n0 + (lane & 15)) * CU_;
  v8f acc = {};
  for (int kc = 0; kc < 8; kc++) {
    v16bf a  = load_a_frag(arow, kc * 32, lane);
    v16bf bf = load_b_frag(brow, kc * 32, lane);
    acc = __builtin_amdgcn_wmma_f32_16x16x32_bf16(false, a, false, bf, (short)0, acc,
                                                  false, false);
  }
  int n = n0 + (lane & 15);
  float b0 = bias[n];
#pragma unroll
  for (int i = 0; i < 8; i++) {
    int m = m0 + i + ((lane >> 4) << 3);
    xpd[(size_t)m * G3 + n] = (__bf16)(acc[i] + b0);
  }
}

// Recurrence: 64 WGs per direction, each owns 8 hidden units (24 gate cols, padded to 32).
// Wr slice LDS-resident; shared h bf16 ping-pong in L2; per-step agent-scope spin barrier.
__global__ void __launch_bounds__(128) k_recur(const __bf16* __restrict__ xp,
                                               const float* __restrict__ fR,
                                               const float* __restrict__ bR,
                                               const float* __restrict__ fb,
                                               const float* __restrict__ bb_,
                                               const float* __restrict__ maskf,
                                               float* hf_g, __bf16* hb_g, int* cnt,
                                               float* out_enc, float* out_states) {
  __shared__ __bf16 ldsW[24 * 512];    // [col(24)][k(512)] Wr slice, bf16
  __shared__ float  ldsHP[32 * 32];    // hp tile 32 batches x 32 cols (24 used)

  int dir = (int)blockIdx.x >> 6;      // 64 WGs per direction
  int wgu = (int)blockIdx.x & 63;
  int u0  = wgu << 3;                  // 8 hidden units per WG
  const float* R  = dir ? bR : fR;     // [GU][G3]
  const float* rb = (dir ? bb_ : fb) + G3;  // bias row 1 (recurrent)
  const __bf16* xpd = xp + (size_t)dir * TOUT * B_ * G3;
  float*  hf  = hf_g + (size_t)dir * B_ * GU_;
  __bf16* hbd = hb_g + (size_t)dir * 2 * B_ * GU_;
  int* c0 = cnt + dir * TOUT;
  int tid = threadIdx.x;

  // Stage this WG's Wr slice into LDS as [c][k], c = gate*8 + u_local.
  for (int i = tid; i < 24 * 512; i += 128) {
    int c = i >> 9, k = i & 511;
    int g = c >> 3, ul = c & 7;
    ldsW[i] = (__bf16)R[(size_t)k * G3 + g * GU_ + u0 + ul];
  }
  __syncthreads();

  int wave = tid >> 5, lane = tid & 31;
  int m0 = (wave >> 1) << 4;           // batch tile: 0 or 16
  int n0 = (wave & 1) << 4;            // col tile:   0 or 16
  int ab = m0 + (lane & 15);           // batch row for A frag
  int bn = n0 + (lane & 15);           // col for B frag / epilogue

  for (int s = 0; s < TOUT; s++) {
    int t = dir ? (TOUT - 1 - s) : s;
    const __bf16* hread = hbd + (size_t)(s & 1) * B_ * GU_;
    __bf16* hwrite      = hbd + (size_t)((s + 1) & 1) * B_ * GU_;
    const __bf16* arow  = hread + (size_t)ab * GU_;

    v8f acc = {};
    for (int kc = 0; kc < 16; kc++) {
      v16bf a = load_a_frag(arow, kc * 32, lane);
      v16bf bf = {};
      if (bn < 24) bf = load_b_frag(ldsW + (size_t)bn * 512, kc * 32, lane);
      acc = __builtin_amdgcn_wmma_f32_16x16x32_bf16(false, a, false, bf, (short)0, acc,
                                                    false, false);
    }
#pragma unroll
    for (int i = 0; i < 8; i++) {
      int row = m0 + i + ((lane >> 4) << 3);
      ldsHP[row * 32 + bn] = acc[i];
    }
    __syncthreads();

    // gate math: 32 batches x 8 units = 256 items over 128 threads
    for (int it = tid; it < 256; it += 128) {
      int b = it >> 3, ul = it & 7;
      int u = u0 + ul;
      const __bf16* xr = xpd + ((size_t)t * B_ + b) * G3;
      float xz  = (float)xr[u];
      float xrr = (float)xr[GU_ + u];
      float xh  = (float)xr[2 * GU_ + u];
      float hz = ldsHP[b * 32 + ul]      + rb[u];
      float hr = ldsHP[b * 32 + 8 + ul]  + rb[GU_ + u];
      float hh = ldsHP[b * 32 + 16 + ul] + rb[2 * GU_ + u];
      float hprev = hf[b * GU_ + u];
      float z = sigmoidf_(xz + hz);
      float r = sigmoidf_(xrr + hr);
      float c = tanhf(xh + r * hh);
      float hn = z * hprev + (1.f - z) * c;
      float mk = maskf[b * TOUT + t];
      hn = (mk > 0.5f) ? hn : hprev;
      hf[b * GU_ + u] = hn;
      hwrite[b * GU_ + u] = (__bf16)hn;
      out_enc[((size_t)b * TOUT + t) * 1024 + dir * GU_ + u] = hn;
      if (s == TOUT - 1) out_states[(size_t)b * 1024 + dir * GU_ + u] = hn;
      // prefetch next step's xp row into cache (global_prefetch_b8)
      int tn = dir ? (t - 1) : (t + 1);
      if (tn >= 0 && tn < TOUT)
        __builtin_prefetch(xpd + ((size_t)tn * B_ + b) * G3 + u, 0, 1);
    }
    __syncthreads();
    __threadfence();
    if (tid == 0) {
      __hip_atomic_fetch_add(c0 + s, 1, __ATOMIC_ACQ_REL, __HIP_MEMORY_SCOPE_AGENT);
      while (__hip_atomic_load(c0 + s, __ATOMIC_ACQUIRE, __HIP_MEMORY_SCOPE_AGENT) < 64)
        __builtin_amdgcn_s_sleep(1);
    }
    __syncthreads();
    __threadfence();
  }
}

// ---------------- launcher ----------------
extern "C" void kernel_launch(void* const* d_in, const int* in_sizes, int n_in,
                              void* d_out, int out_size, void* d_ws, size_t ws_size,
                              hipStream_t stream) {
  const float* x      = (const float*)d_in[0];
  const float* conv_k = (const float*)d_in[1];
  const float* conv_b = (const float*)d_in[2];
  const float* fwd_K  = (const float*)d_in[3];
  const float* fwd_R  = (const float*)d_in[4];
  const float* fwd_b  = (const float*)d_in[5];
  const float* bwd_K  = (const float*)d_in[6];
  const float* bwd_R  = (const float*)d_in[7];
  const float* bwd_b  = (const float*)d_in[8];

  char* ws = (char*)d_ws;
  __bf16* y   = (__bf16*)(ws + WS_Y);
  __bf16* xp  = (__bf16*)(ws + WS_XP);
  __bf16* wT  = (__bf16*)(ws + WS_WTCONV);
  __bf16* kT  = (__bf16*)(ws + WS_KT);
  float*  hf  = (float*)(ws + WS_HF);
  __bf16* hb  = (__bf16*)(ws + WS_HB);
  float*  mk  = (float*)(ws + WS_MASK);
  int*    cnt = (int*)(ws + WS_CNT);

  float* out        = (float*)d_out;
  float* out_enc    = out;                                   // [B][TOUT][1024]
  float* out_states = out + (size_t)B_ * TOUT * 1024;        // [B][1024]
  float* out_mask   = out_states + (size_t)B_ * 1024;        // [B][TOUT]

  k_init<<<256, 256, 0, stream>>>(hf, hb, cnt);
  k_prep<<<512, 256, 0, stream>>>(conv_k, fwd_K, bwd_K, wT, kT);
  k_mask<<<64, 256, 0, stream>>>(x, mk, out_mask);
  k_conv<<<1024, 256, 0, stream>>>(x, wT, conv_b, y);
  dim3 pg(12288, 2);
  k_proj<<<pg, 256, 0, stream>>>(y, kT, fwd_b, bwd_b, xp);
  k_recur<<<128, 128, 0, stream>>>(xp, fwd_R, bwd_R, fwd_b, bwd_b, mk,
                                   hf, hb, cnt, out_enc, out_states);
}
